// MLA_60481729462383
// MI455X (gfx1250) — compile-verified, matrix operand order checked by reading, not run on hardware
//
#include <hip/hip_runtime.h>
#include <hip/hip_bf16.h>

typedef __bf16 bf16;
typedef __attribute__((ext_vector_type(16))) __bf16 v16bf;
typedef __attribute__((ext_vector_type(8)))  __bf16 v8bf;
typedef __attribute__((ext_vector_type(8)))  float  v8f;

#define D_MODEL 2048
#define N_HEADS 16
#define D_HEAD  128
#define D_ROPE  64
#define D_KV    512
#define B_      2
#define S_      2048
#define M_ROWS  (B_ * S_)   // 4096

__device__ __forceinline__ v8f wmma_bf16(v16bf a, v16bf b, v8f c) {
  return __builtin_amdgcn_wmma_f32_16x16x32_bf16(
      false, a, false, b, (short)0, c, false, false);
}

__device__ __forceinline__ v16bf cat16(v8bf lo, v8bf hi) {
  v16bf r;
#pragma unroll
  for (int i = 0; i < 8; ++i) { r[i] = lo[i]; r[i + 8] = hi[i]; }
  return r;
}

// Async global->LDS 16-byte DMA (tracked by ASYNCcnt, §15.18.3 op 98).
__device__ __forceinline__ void async_copy_b128(unsigned lds_off, const bf16* g) {
  asm volatile("global_load_async_to_lds_b128 %0, %1, off"
               :: "v"(lds_off), "v"((unsigned long long)(size_t)g)
               : "memory");
}

// ---------------------------------------------------------------------------
// GEMM: C[M,N] = A[M,K](bf16) @ WT[N,K](bf16, K-major) + bias[N]
// Outputs: optional f32 C and/or bf16 C.
// Block 256 = 8 waves; tile 128(M) x 64(N); K-step 64; double-buffered LDS
// filled by GLOBAL_LOAD_ASYNC_TO_LDS_B128 (6 x 16B per thread per tile),
// s_wait_asynccnt ping-pong. Wave (wm,wn) computes 32x32 (8 WMMA / K-step).
// Requires M%128==0, N%64==0, K%64==0 (true for all call sites).
// ---------------------------------------------------------------------------
__global__ __launch_bounds__(256)
void gemm_bf16_wmma(const bf16* __restrict__ A, const bf16* __restrict__ WT,
                    const float* __restrict__ bias,
                    float* __restrict__ Cf, bf16* __restrict__ Cbf,
                    int M, int K, int N) {
  __shared__ bf16 As[2][128][64];   // [m][k]
  __shared__ bf16 Bs[2][64][64];    // [n][k]

  const int tid  = threadIdx.x;
  const int lane = tid & 31;
  const int wid  = tid >> 5;
  const int wm   = wid & 3;         // 4 m-blocks of 32 rows
  const int wn   = wid >> 2;        // 2 n-blocks of 32 cols
  const int lh   = lane & 15;
  const int kg   = lane >> 4;
  const int m0 = blockIdx.y * 128;
  const int n0 = blockIdx.x * 64;

  const int ar = tid >> 3;          // 0..31: base row for copies
  const int ac = (tid & 7) * 8;     // col (bf16 elems), 16B granules

  v8f acc[2][2];
#pragma unroll
  for (int i = 0; i < 2; ++i)
#pragma unroll
    for (int j = 0; j < 2; ++j) acc[i][j] = (v8f){};

  auto issue = [&](int k0, int buf) {
    const bf16* ag = A + (size_t)(m0 + ar) * K + k0 + ac;
    unsigned al = (unsigned)(size_t)&As[buf][ar][ac];
#pragma unroll
    for (int j = 0; j < 4; ++j)                       // rows ar, ar+32, +64, +96
      async_copy_b128(al + (unsigned)(j * 32 * 64 * 2), ag + (size_t)(32 * j) * K);
    const bf16* bg = WT + (size_t)(n0 + ar) * K + k0 + ac;
    unsigned bl = (unsigned)(size_t)&Bs[buf][ar][ac];
#pragma unroll
    for (int j = 0; j < 2; ++j)                       // rows ar, ar+32
      async_copy_b128(bl + (unsigned)(j * 32 * 64 * 2), bg + (size_t)(32 * j) * K);
  };

  issue(0, 0);
  const int nk = K >> 6;
  for (int i = 0; i < nk; ++i) {
    const int buf = i & 1;
    if (i + 1 < nk) {
      issue((i + 1) << 6, buf ^ 1);
      asm volatile("s_wait_asynccnt 0x6" ::: "memory");  // prev tile's 6 done
    } else {
      asm volatile("s_wait_asynccnt 0x0" ::: "memory");
    }
    __syncthreads();
#pragma unroll
    for (int c = 0; c < 2; ++c) {
      const int kc = 32 * c;
      v16bf a0 = cat16(*(const v8bf*)&As[buf][32 * wm + lh][kc + kg * 8],
                       *(const v8bf*)&As[buf][32 * wm + lh][kc + 16 + kg * 8]);
      v16bf a1 = cat16(*(const v8bf*)&As[buf][32 * wm + 16 + lh][kc + kg * 8],
                       *(const v8bf*)&As[buf][32 * wm + 16 + lh][kc + 16 + kg * 8]);
      v16bf b0 = cat16(*(const v8bf*)&Bs[buf][32 * wn + lh][kc + kg * 8],
                       *(const v8bf*)&Bs[buf][32 * wn + lh][kc + 16 + kg * 8]);
      v16bf b1 = cat16(*(const v8bf*)&Bs[buf][32 * wn + 16 + lh][kc + kg * 8],
                       *(const v8bf*)&Bs[buf][32 * wn + 16 + lh][kc + 16 + kg * 8]);
      acc[0][0] = wmma_bf16(a0, b0, acc[0][0]);
      acc[0][1] = wmma_bf16(a0, b1, acc[0][1]);
      acc[1][0] = wmma_bf16(a1, b0, acc[1][0]);
      acc[1][1] = wmma_bf16(a1, b1, acc[1][1]);
    }
    __syncthreads();
  }

#pragma unroll
  for (int mb = 0; mb < 2; ++mb)
#pragma unroll
    for (int nb = 0; nb < 2; ++nb) {
      const int n = n0 + 32 * wn + 16 * nb + lh;
      const float bn = bias[n];
#pragma unroll
      for (int r = 0; r < 8; ++r) {
        const int m = m0 + 32 * wm + 16 * mb + r + 8 * kg;
        const float v = acc[mb][nb][r] + bn;
        if (Cf)  Cf[(size_t)m * N + n] = v;
        if (Cbf) Cbf[(size_t)m * N + n] = (bf16)v;
      }
    }
}

// ---------------------------------------------------------------------------
// One-time packers (run per launch; weights converted once, not per tile).
// ---------------------------------------------------------------------------
__global__ void pack_bf16(const float* __restrict__ x, bf16* __restrict__ y) {
  const int i = blockIdx.x * blockDim.x + threadIdx.x;
  y[i] = (bf16)x[i];
}

// W[K,N] f32 -> WT[N,K] bf16
__global__ void pack_wT(const float* __restrict__ W, bf16* __restrict__ WT,
                        int K, int N) {
  const int idx = blockIdx.x * blockDim.x + threadIdx.x;  // over N*K
  const int n = idx / K;
  const int k = idx - n * K;
  WT[idx] = (bf16)W[(size_t)k * N + n];
}

// ---------------------------------------------------------------------------
// RoPE: inv_freq[i] = 10000^(-2i/64); rotate_half(x)[d] = d<32 ? -x[d+32] : x[d-32]
// ---------------------------------------------------------------------------
__device__ __forceinline__ float rope_val(const float* rot_row, int dp, int s) {
  const int i = dp & 31;
  const float ang = (float)s * __expf(-(float)i * 0.03125f * 9.210340371976184f);
  float sn, cs;
  __sincosf(ang, &sn, &cs);
  const float x = rot_row[dp];
  const float other = (dp < 32) ? -rot_row[dp + 32] : rot_row[dp - 32];
  return x * cs + other * sn;
}

__global__ void rope_pack_q(const float* __restrict__ qbase,
                            const float* __restrict__ qrot,
                            bf16* __restrict__ qout) {
  const int idx = blockIdx.x * blockDim.x + threadIdx.x;  // B*H*S*128
  const int d = idx & 127;
  const int s = (idx >> 7) & (S_ - 1);
  const int h = (idx >> 18) & (N_HEADS - 1);
  const int b = idx >> 22;
  const int row = b * S_ + s;
  float val;
  if (d < 64) val = qbase[(size_t)row * 1024 + h * 64 + d];
  else        val = rope_val(&qrot[(size_t)row * 1024 + h * 64], d - 64, s);
  qout[idx] = (bf16)val;
}

__global__ void rope_pack_k(const float* __restrict__ kbase,
                            const float* __restrict__ krot,
                            bf16* __restrict__ kout,
                            float* __restrict__ krot_out) {
  const int idx = blockIdx.x * blockDim.x + threadIdx.x;
  const int d = idx & 127;
  const int s = (idx >> 7) & (S_ - 1);
  const int h = (idx >> 18) & (N_HEADS - 1);
  const int b = idx >> 22;
  const int row = b * S_ + s;
  float val;
  if (d < 64) {
    val = kbase[(size_t)row * 1024 + h * 64 + d];
  } else {
    const int dp = d - 64;
    val = rope_val(&krot[(size_t)row * 1024 + h * 64], dp, s);
    krot_out[((size_t)row * N_HEADS + h) * 64 + dp] = val;
  }
  kout[idx] = (bf16)val;
}

// v f32 [B,S,H*128] -> vT bf16 [B,H,128,S]
__global__ void pack_vT(const float* __restrict__ v, bf16* __restrict__ vT) {
  const int idx = blockIdx.x * blockDim.x + threadIdx.x;  // B*H*128*S
  const int s = idx & (S_ - 1);
  const int d = (idx >> 11) & 127;
  const int h = (idx >> 18) & (N_HEADS - 1);
  const int b = idx >> 22;
  vT[idx] = (bf16)v[((size_t)(b * S_ + s)) * 2048 + h * 128 + d];
}

// ---------------------------------------------------------------------------
// Flash attention: one wave per (b, h, 16 query rows); 32-key tiles; online
// softmax via 16-lane shuffle reductions; WMMA for Q.K^T and P.V; bf16 out.
// ---------------------------------------------------------------------------
__global__ __launch_bounds__(32)
void flash_attn_wmma(const bf16* __restrict__ q, const bf16* __restrict__ k,
                     const bf16* __restrict__ vT, bf16* __restrict__ out) {
  __shared__ bf16 P[16][32];   // probabilities: C-layout -> A-layout staging

  const int lane = threadIdx.x;
  const int lh = lane & 15;
  const int kg = lane >> 4;
  const int q0 = blockIdx.x * 16;
  const int h  = blockIdx.y;
  const int b  = blockIdx.z;
  const size_t bh = (size_t)(b * N_HEADS + h);

  const bf16* qp = q + (bh * S_ + q0 + lh) * 128;
  v16bf qf[4];
#pragma unroll
  for (int c = 0; c < 4; ++c)
    qf[c] = cat16(*(const v8bf*)(qp + 32 * c + kg * 8),
                  *(const v8bf*)(qp + 32 * c + 16 + kg * 8));

  v8f o[8];
#pragma unroll
  for (int a = 0; a < 8; ++a) o[a] = (v8f){};
  float mrow[8], lrow[8];
#pragma unroll
  for (int r = 0; r < 8; ++r) { mrow[r] = -3.0e38f; lrow[r] = 0.0f; }
  const float scale = 0.08838834764831845f;  // 1/sqrt(128)

  for (int key0 = 0; key0 < S_; key0 += 32) {
    v8f s0 = {};
    v8f s1 = {};
    const bf16* kp0 = k + (bh * S_ + key0 + lh) * 128;
    const bf16* kp1 = kp0 + 16 * 128;
#pragma unroll
    for (int c = 0; c < 4; ++c) {
      v16bf kb = cat16(*(const v8bf*)(kp0 + 32 * c + kg * 8),
                       *(const v8bf*)(kp0 + 32 * c + 16 + kg * 8));
      s0 = wmma_bf16(qf[c], kb, s0);
    }
#pragma unroll
    for (int c = 0; c < 4; ++c) {
      v16bf kb = cat16(*(const v8bf*)(kp1 + 32 * c + kg * 8),
                       *(const v8bf*)(kp1 + 32 * c + 16 + kg * 8));
      s1 = wmma_bf16(qf[c], kb, s1);
    }

    float alph[8];
#pragma unroll
    for (int r = 0; r < 8; ++r) {
      const float a0 = s0[r] * scale;
      const float a1 = s1[r] * scale;
      float rm = fmaxf(a0, a1);
#pragma unroll
      for (int off = 1; off < 16; off <<= 1) rm = fmaxf(rm, __shfl_xor(rm, off, 32));
      const float mnew = fmaxf(mrow[r], rm);
      const float p0 = __expf(a0 - mnew);
      const float p1 = __expf(a1 - mnew);
      float rs = p0 + p1;
#pragma unroll
      for (int off = 1; off < 16; off <<= 1) rs += __shfl_xor(rs, off, 32);
      alph[r] = __expf(mrow[r] - mnew);
      lrow[r] = lrow[r] * alph[r] + rs;
      mrow[r] = mnew;
      P[r + 8 * kg][lh]      = (bf16)p0;
      P[r + 8 * kg][16 + lh] = (bf16)p1;
    }
#pragma unroll
    for (int a = 0; a < 8; ++a)
#pragma unroll
      for (int r = 0; r < 8; ++r) o[a][r] *= alph[r];
    __syncthreads();

    v16bf pf = cat16(*(const v8bf*)&P[lh][kg * 8],
                     *(const v8bf*)&P[lh][16 + kg * 8]);
    const bf16* vp = vT + bh * 128 * S_ + key0;
#pragma unroll
    for (int a = 0; a < 8; ++a) {
      const bf16* vpp = vp + (size_t)(16 * a + lh) * S_;
      v16bf vb = cat16(*(const v8bf*)(vpp + kg * 8),
                       *(const v8bf*)(vpp + 16 + kg * 8));
      o[a] = wmma_bf16(pf, vb, o[a]);
    }
    __syncthreads();
  }

#pragma unroll
  for (int r = 0; r < 8; ++r) {
    const float inv = 1.0f / lrow[r];
    const int srow = q0 + r + 8 * kg;
    bf16* op = out + ((size_t)(b * S_ + srow)) * 2048 + h * 128;
#pragma unroll
    for (int a = 0; a < 8; ++a) op[16 * a + lh] = (bf16)(o[a][r] * inv);
  }
}

// ---------------------------------------------------------------------------
extern "C" void kernel_launch(void* const* d_in, const int* in_sizes, int n_in,
                              void* d_out, int out_size, void* d_ws, size_t ws_size,
                              hipStream_t stream) {
  (void)in_sizes; (void)n_in; (void)out_size; (void)ws_size;

  const float* h     = (const float*)d_in[0];
  const float* W_dkv = (const float*)d_in[1];
  const float* b_dkv = (const float*)d_in[2];
  const float* W_dq  = (const float*)d_in[3];
  const float* b_dq  = (const float*)d_in[4];
  const float* W_uk  = (const float*)d_in[5];
  const float* b_uk  = (const float*)d_in[6];
  const float* W_uv  = (const float*)d_in[7];
  const float* b_uv  = (const float*)d_in[8];
  const float* W_uq  = (const float*)d_in[9];
  const float* b_uq  = (const float*)d_in[10];
  const float* W_qr  = (const float*)d_in[11];
  const float* b_qr  = (const float*)d_in[12];
  const float* W_kr  = (const float*)d_in[13];
  const float* b_kr  = (const float*)d_in[14];
  const float* W_o   = (const float*)d_in[15];
  const float* b_o   = (const float*)d_in[16];

  float* out_final = (float*)d_out;                          // [B,S,2048]
  float* out_ckv   = out_final + (size_t)M_ROWS * D_MODEL;   // [B,S,512]
  float* out_krot  = out_ckv + (size_t)M_ROWS * D_KV;        // [B,S,H,64]

  char* ws = (char*)d_ws;
  bf16*  hbf    = (bf16*)(ws + (0ull   << 20));  // 16 MiB [4096,2048]
  bf16*  wTdkv  = (bf16*)(ws + (16ull  << 20));  //  2 MiB [512,2048]
  bf16*  wTdq   = (bf16*)(ws + (18ull  << 20));  //  2 MiB [512,2048]
  bf16*  wTuk   = (bf16*)(ws + (20ull  << 20));  //  1 MiB [1024,512]
  bf16*  wTuv   = (bf16*)(ws + (21ull  << 20));  //  2 MiB [2048,512]
  bf16*  wTuq   = (bf16*)(ws + (23ull  << 20));  //  1 MiB [1024,512]
  bf16*  wTqr   = (bf16*)(ws + (24ull  << 20));  //  1 MiB [1024,512]
  bf16*  wTkr   = (bf16*)(ws + (25ull  << 20));  //  4 MiB [1024,2048]
  bf16*  wTo    = (bf16*)(ws + (29ull  << 20));  //  8 MiB [2048,2048]
  bf16*  ckv_bf = (bf16*)(ws + (37ull  << 20));  //  4 MiB [4096,512]
  bf16*  cq_bf  = (bf16*)(ws + (41ull  << 20));  //  4 MiB [4096,512]
  float* kbase  = (float*)(ws + (45ull << 20));  // 16 MiB [4096,1024]
  float* qbase  = (float*)(ws + (61ull << 20));  // 16 MiB [4096,1024]
  float* qrot   = (float*)(ws + (77ull << 20));  // 16 MiB [4096,1024]
  float* krot   = (float*)(ws + (93ull << 20));  // 16 MiB [4096,1024]
  float* vbuf   = (float*)(ws + (109ull << 20)); // 32 MiB [4096,2048]
  bf16*  attnbf = (bf16*)(ws + (141ull << 20));  // 16 MiB [4096,2048]
  bf16*  qbf    = (bf16*)(ws + (157ull << 20));  // 16 MiB [B,H,S,128]
  bf16*  kbf    = (bf16*)(ws + (173ull << 20));  // 16 MiB [B,H,S,128]
  bf16*  vTbf   = (bf16*)(ws + (189ull << 20));  // 16 MiB [B,H,128,S]

  // --- one-time bf16 packing ---
  pack_bf16<<<(M_ROWS * D_MODEL) / 256, 256, 0, stream>>>(h, hbf);
  pack_wT<<<(2048 * 512)  / 256, 256, 0, stream>>>(W_dkv, wTdkv, 2048, 512);
  pack_wT<<<(2048 * 512)  / 256, 256, 0, stream>>>(W_dq,  wTdq,  2048, 512);
  pack_wT<<<(512 * 1024)  / 256, 256, 0, stream>>>(W_uk,  wTuk,  512, 1024);
  pack_wT<<<(512 * 2048)  / 256, 256, 0, stream>>>(W_uv,  wTuv,  512, 2048);
  pack_wT<<<(512 * 1024)  / 256, 256, 0, stream>>>(W_uq,  wTuq,  512, 1024);
  pack_wT<<<(512 * 1024)  / 256, 256, 0, stream>>>(W_qr,  wTqr,  512, 1024);
  pack_wT<<<(2048 * 1024) / 256, 256, 0, stream>>>(W_kr,  wTkr,  2048, 1024);
  pack_wT<<<(2048 * 2048) / 256, 256, 0, stream>>>(W_o,   wTo,   2048, 2048);

  const dim3 blk(256);
  // --- projections (all WMMA, async-LDS pipelined) ---
  gemm_bf16_wmma<<<dim3(512 / 64,  M_ROWS / 128), blk, 0, stream>>>(hbf,    wTdkv, b_dkv, out_ckv, ckv_bf, M_ROWS, 2048, 512);
  gemm_bf16_wmma<<<dim3(512 / 64,  M_ROWS / 128), blk, 0, stream>>>(hbf,    wTdq,  b_dq,  nullptr, cq_bf,  M_ROWS, 2048, 512);
  gemm_bf16_wmma<<<dim3(1024 / 64, M_ROWS / 128), blk, 0, stream>>>(ckv_bf, wTuk,  b_uk,  kbase,   nullptr, M_ROWS, 512, 1024);
  gemm_bf16_wmma<<<dim3(2048 / 64, M_ROWS / 128), blk, 0, stream>>>(ckv_bf, wTuv,  b_uv,  vbuf,    nullptr, M_ROWS, 512, 2048);
  gemm_bf16_wmma<<<dim3(1024 / 64, M_ROWS / 128), blk, 0, stream>>>(cq_bf,  wTuq,  b_uq,  qbase,   nullptr, M_ROWS, 512, 1024);
  gemm_bf16_wmma<<<dim3(1024 / 64, M_ROWS / 128), blk, 0, stream>>>(cq_bf,  wTqr,  b_qr,  qrot,    nullptr, M_ROWS, 512, 1024);
  gemm_bf16_wmma<<<dim3(1024 / 64, M_ROWS / 128), blk, 0, stream>>>(hbf,    wTkr,  b_kr,  krot,    nullptr, M_ROWS, 2048, 1024);

  // --- RoPE + attention-layout packing ---
  const int nel = B_ * N_HEADS * S_ * D_HEAD;  // 8388608
  rope_pack_q<<<nel / 256, 256, 0, stream>>>(qbase, qrot, qbf);
  rope_pack_k<<<nel / 256, 256, 0, stream>>>(kbase, krot, kbf, out_krot);
  pack_vT<<<nel / 256, 256, 0, stream>>>(vbuf, vTbf);

  // --- attention ---
  flash_attn_wmma<<<dim3(S_ / 16, N_HEADS, B_), dim3(32), 0, stream>>>(qbf, kbf, vTbf, attnbf);

  // --- output projection ---
  gemm_bf16_wmma<<<dim3(2048 / 64, M_ROWS / 128), blk, 0, stream>>>(attnbf, wTo, b_o, out_final, nullptr, M_ROWS, 2048, 2048);
}
